// LSTM_CRF_49718541418938
// MI455X (gfx1250) — compile-verified, matrix operand order checked by reading, not run on hardware
//
#include <hip/hip_runtime.h>
#include <hip/hip_bf16.h>
#include <stdint.h>

// ---------------- problem constants (from reference) ----------------
constexpr int E_SZ = 256;
constexpr int H_SZ = 512;
constexpr int HD   = 256;      // per-direction hidden
constexpr int G4   = 1024;     // 4*HD gate width
constexpr int KT   = 20;       // tags
constexpr int B_SZ = 64;
constexpr int T_SZ = 512;
constexpr int START_TAG = 17;
constexpr int END_TAG   = 18;
constexpr float NEGV = -10000.0f;
constexpr int BT = B_SZ * T_SZ;   // 32768 rows

typedef __bf16 bf16_t;
typedef __attribute__((ext_vector_type(2)))  __bf16 bf16x2;
typedef __attribute__((ext_vector_type(16))) __bf16 v16bf;
typedef __attribute__((ext_vector_type(8)))  float  v8f;
typedef __attribute__((ext_vector_type(4)))  unsigned int u32x4;
typedef __attribute__((ext_vector_type(8)))  int i32x8;
typedef __attribute__((ext_vector_type(4)))  int i32x4;

// ---------------- TDM (Tensor Data Mover) support, gfx1250 ----------------
#if defined(__has_builtin)
#  if __has_builtin(__builtin_amdgcn_tensor_load_to_lds) && \
      __has_builtin(__builtin_amdgcn_s_wait_tensorcnt)
#    define HAVE_TDM 1
#  endif
#endif
#ifndef HAVE_TDM
#  define HAVE_TDM 0
#endif

#if HAVE_TDM
// 2D tile load Global->LDS via one TDM descriptor (D# groups per ISA 8.3/8.4).
// This toolchain exposes the 6-arg builtin:
//   (u32x4 g0, i32x8 g1, i32x4 g2, i32x4 g3, i32x8 extra, i32 cpol)
// Groups 2/3 (+extra) are zero: 2D tensor, higher dims unused.
// dsz_code: 0=1B,1=2B,2=4B,3=8B.  pad_int: 2^(v+1) dwords before each pad;
// pad_amt: (v+1) dwords of pad inserted into the LDS stream (load only).
__device__ __forceinline__ void tdm_load_2d(const void* gsrc, uint32_t lds_off,
                                            uint32_t td0, uint32_t td1,
                                            uint64_t stride0_elems,
                                            uint32_t tile0, uint32_t tile1,
                                            uint32_t dsz_code, bool pad_en,
                                            uint32_t pad_int, uint32_t pad_amt) {
  const uint64_t ga = (uint64_t)(uintptr_t)gsrc;
  u32x4 g0;
  g0[0] = 1u;                                    // count=1, user mode, no gather
  g0[1] = lds_off;                               // LDS byte address
  g0[2] = (uint32_t)ga;                          // global_addr[31:0]
  g0[3] = (uint32_t)((ga >> 32) & 0x1FFFFFFu) | (2u << 30);  // [56:32] | type=2
  i32x8 g1;
  g1[0] = (int)((dsz_code << 16) | ((pad_en ? 1u : 0u) << 20) |
                (pad_int << 22) | (pad_amt << 25));          // mask=0,flags
  g1[1] = (int)((td0 & 0xFFFFu) << 16);          // tensor_dim0[15:0] @ bit48
  g1[2] = (int)(((td0 >> 16) & 0xFFFFu) | ((td1 & 0xFFFFu) << 16));
  g1[3] = (int)(((td1 >> 16) & 0xFFFFu) | ((tile0 & 0xFFFFu) << 16));
  g1[4] = (int)(tile1 & 0xFFFFu);                // tile_dim1, tile_dim2=0
  g1[5] = (int)(uint32_t)(stride0_elems & 0xFFFFFFFFu);
  g1[6] = (int)(uint32_t)((stride0_elems >> 32) & 0xFFFFu);  // dim1_stride=0
  g1[7] = 0;
  i32x4 gz4;
  gz4[0] = 0; gz4[1] = 0; gz4[2] = 0; gz4[3] = 0;
  i32x8 gz8;
#pragma unroll
  for (int i = 0; i < 8; ++i) gz8[i] = 0;
  __builtin_amdgcn_tensor_load_to_lds(g0, g1, gz4, gz4, gz8, 0);
}
#endif

// ---------------- WMMA fragment helpers (ISA 7.12.2 layouts) ----------------
__device__ __forceinline__ v16bf make_frag(const bf16_t* base, int pitch,
                                           int rowBase, int kk) {
  const int lane  = threadIdx.x & 31;
  const int hf    = lane >> 4;
  const int row   = rowBase + (lane & 15);
  const bf16_t* p = base + (size_t)row * pitch + kk;
  v16bf f;
#pragma unroll
  for (int v = 0; v < 8; ++v) {
    const int ko = (v < 4) ? (hf * 8 + 2 * v) : (16 + hf * 8 + 2 * (v - 4));
    bf16x2 pr = *reinterpret_cast<const bf16x2*>(p + ko);
    f[2 * v]     = pr.x;
    f[2 * v + 1] = pr.y;
  }
  return f;
}

__device__ __forceinline__ float sigmoidf_(float x) {
  return 1.0f / (1.0f + __expf(-x));
}

// ---------------- prep kernels ----------------
__global__ void k_embed_gather(const int* __restrict__ x,
                               const float* __restrict__ embed,
                               bf16_t* __restrict__ emb_bf) {
  size_t i = (size_t)blockIdx.x * blockDim.x + threadIdx.x;
  if (i >= (size_t)BT * E_SZ) return;
  int row = (int)(i >> 8);          // E=256
  int e   = (int)(i & 255);
  int tok = x[row];
  emb_bf[i] = (bf16_t)embed[(size_t)tok * E_SZ + e];
}

__global__ void k_f32_to_bf16(const float* __restrict__ src,
                              bf16_t* __restrict__ dst, int n) {
  int i = blockIdx.x * blockDim.x + threadIdx.x;
  if (i < n) dst[i] = (bf16_t)src[i];
}

__global__ void k_bias_fuse(const float* __restrict__ bih,
                            const float* __restrict__ bhh,
                            float* __restrict__ bias, int n) {
  int i = blockIdx.x * blockDim.x + threadIdx.x;
  if (i < n) bias[i] = bih[i] + bhh[i];
}

// ---------------- G = emb(BTx256) @ Wih^T(256x1024), bf16 WMMA ----------------
// block tile 128(M) x 64(N), K step 32; 8 waves, each 32x32 (2x2 WMMA tiles).
// A tile staged by the Tensor Data Mover (row pad 16B -> As[128][40] layout);
// B tile staged by vector loads (keeps LOADcnt path warm for comparison).
__global__ __launch_bounds__(256, 2)
void k_gemm_xproj(const bf16_t* __restrict__ A,     // [BT][256]
                  const bf16_t* __restrict__ Wih,   // [1024][256] row-major
                  bf16_t* __restrict__ G) {         // [BT][1024]
  constexpr int BKp = 32 + 8;                       // padded K pitch
  __shared__ bf16_t As[128][BKp];
  __shared__ bf16_t Bs[64][BKp];
  const int tid  = threadIdx.x;
  const int wave = tid >> 5, lane = tid & 31;
  const int lane15 = lane & 15, hf = lane >> 4;
  const int bm = blockIdx.x * 128, bn = blockIdx.y * 64;
  const int mw = (wave >> 1) * 32, nw = (wave & 1) * 32;

  v8f acc[2][2];
#pragma unroll
  for (int i = 0; i < 2; ++i)
#pragma unroll
    for (int j = 0; j < 2; ++j) { v8f z = {}; acc[i][j] = z; }

  for (int kk = 0; kk < 256; kk += 32) {
#if HAVE_TDM
    // one TDM descriptor moves the whole 128x32 bf16 A tile, padding each
    // 64B row (16 dwords, code 3) with 16B (4 dwords, code 3) -> pitch 40.
    if (wave == 0) {
      tdm_load_2d(A + (size_t)bm * 256 + kk,
                  (uint32_t)(uintptr_t)&As[0][0],
                  /*tensor_d0=*/256, /*tensor_d1=*/(uint32_t)BT,
                  /*stride0=*/256, /*tile0=*/32, /*tile1=*/128,
                  /*dsz=*/1, /*pad=*/true, /*pad_int=*/3, /*pad_amt=*/3);
    }
#else
#pragma unroll
    for (int i = 0; i < 2; ++i) {
      int idx = tid + i * 256;
      int r = idx >> 2, q = idx & 3;
      const uint4* s =
          reinterpret_cast<const uint4*>(A + (size_t)(bm + r) * 256 + kk + q * 8);
      *reinterpret_cast<uint4*>(&As[r][q * 8]) = *s;
    }
#endif
    // stage B tile: 64x32 bf16 (256 uint4), 1 per thread
    {
      int r = tid >> 2, q = tid & 3;
      const uint4* s =
          reinterpret_cast<const uint4*>(Wih + (size_t)(bn + r) * 256 + kk + q * 8);
      *reinterpret_cast<uint4*>(&Bs[r][q * 8]) = *s;
    }
    if (kk + 32 < 256) {  // gfx1250 global_prefetch_b8 for next K tile
      __builtin_prefetch(A + (size_t)(bm + (tid >> 1)) * 256 + kk + 32, 0, 0);
      __builtin_prefetch(Wih + (size_t)(bn + (tid & 63)) * 256 + kk + 32, 0, 0);
    }
#if HAVE_TDM
    if (wave == 0) __builtin_amdgcn_s_wait_tensorcnt(0);
#endif
    __syncthreads();

    v16bf a0 = make_frag(&As[0][0], BKp, mw, 0);
    v16bf a1 = make_frag(&As[0][0], BKp, mw + 16, 0);
    v16bf b0 = make_frag(&Bs[0][0], BKp, nw, 0);
    v16bf b1 = make_frag(&Bs[0][0], BKp, nw + 16, 0);
    acc[0][0] = __builtin_amdgcn_wmma_f32_16x16x32_bf16(false, a0, false, b0,
                                                        (short)0, acc[0][0], false, false);
    acc[0][1] = __builtin_amdgcn_wmma_f32_16x16x32_bf16(false, a0, false, b1,
                                                        (short)0, acc[0][1], false, false);
    acc[1][0] = __builtin_amdgcn_wmma_f32_16x16x32_bf16(false, a1, false, b0,
                                                        (short)0, acc[1][0], false, false);
    acc[1][1] = __builtin_amdgcn_wmma_f32_16x16x32_bf16(false, a1, false, b1,
                                                        (short)0, acc[1][1], false, false);
    __syncthreads();
  }

  // C/D layout: vgpr r -> local M = r + 8*half ; N = lane&15
#pragma unroll
  for (int tm = 0; tm < 2; ++tm)
#pragma unroll
    for (int tn = 0; tn < 2; ++tn)
#pragma unroll
      for (int r = 0; r < 8; ++r) {
        int row = bm + mw + tm * 16 + r + 8 * hf;
        int col = bn + nw + tn * 16 + lane15;
        G[(size_t)row * G4 + col] = (bf16_t)acc[tm][tn][r];
      }
}

// ---------------- persistent recurrent LSTM (one block per direction) --------
// gates(64x1024) = G[t] + bias + h(64x256) @ Whh^T ; WMMA bf16, f32 accum.
// The 64x1024 bf16 G[t] tile (rows strided by T*1024) is streamed into LDS by
// the TDM, double-buffered: step s+1's tile DMAs while step s computes.
constexpr int HP = HD + 4;                           // padded h pitch
constexpr size_t SM_HBUF  = (size_t)B_SZ * HP * 2;   // 33280 B
constexpr size_t SM_GTILE = (size_t)2 * B_SZ * G4 * 2;  // 262144 B
constexpr size_t SM_RECUR = SM_HBUF + SM_GTILE;      // ~289 KB of 320 KB WGP LDS

__global__ __launch_bounds__(1024, 1)
void k_lstm_recurrent(const bf16_t* __restrict__ G_f,
                      const bf16_t* __restrict__ G_b,
                      const bf16_t* __restrict__ Whh_f,
                      const bf16_t* __restrict__ Whh_b,
                      const float* __restrict__ bias_f,
                      const float* __restrict__ bias_b,
                      float* __restrict__ c_ws,     // [2][64*256]
                      bf16_t* __restrict__ h_out) { // [BT][512]
  extern __shared__ char dynsmem[];
  bf16_t* hbuf = (bf16_t*)dynsmem;                  // [64][HP]
#if HAVE_TDM
  bf16_t* gtile = (bf16_t*)(dynsmem + SM_HBUF);     // [2][64][1024]
#endif
  const int dir = blockIdx.x;
  const bf16_t* G    = dir ? G_b : G_f;
  const bf16_t* Whh  = dir ? Whh_b : Whh_f;
  const float*  bias = dir ? bias_b : bias_f;
  float* cbuf = c_ws + (size_t)dir * B_SZ * HD;

  const int tid = threadIdx.x;
  const int wave = tid >> 5, lane = tid & 31;
  const int lane15 = lane & 15, hf = lane >> 4;

  for (int i = tid; i < B_SZ * HP; i += 1024) hbuf[i] = (bf16_t)0.0f;
  for (int i = tid; i < B_SZ * HD; i += 1024) cbuf[i] = 0.0f;
  __syncthreads();

#if HAVE_TDM
  if (wave == 0) {
    const int t0 = dir ? (T_SZ - 1) : 0;
    tdm_load_2d(G + (size_t)t0 * G4, (uint32_t)(uintptr_t)gtile,
                /*tensor_d0=*/G4, /*tensor_d1=*/B_SZ,
                /*stride0=*/(uint64_t)T_SZ * G4,
                /*tile0=*/G4, /*tile1=*/B_SZ,
                /*dsz=*/1, /*pad=*/false, 0, 0);
  }
#endif

  float hreg[2][8];
  for (int s = 0; s < T_SZ; ++s) {
    const int t = dir ? (T_SZ - 1 - s) : s;
#if HAVE_TDM
    if (wave == 0) __builtin_amdgcn_s_wait_tensorcnt(0);
    __syncthreads();                     // tile s ready for all waves
    if (s + 1 < T_SZ && wave == 0) {     // prefetch tile s+1 into other buffer
      const int tn = dir ? (T_SZ - 2 - s) : (s + 1);
      tdm_load_2d(G + (size_t)tn * G4,
                  (uint32_t)(uintptr_t)(gtile + (size_t)((s + 1) & 1) * B_SZ * G4),
                  G4, B_SZ, (uint64_t)T_SZ * G4, G4, B_SZ, 1, false, 0, 0);
    }
    const bf16_t* gt = gtile + (size_t)(s & 1) * B_SZ * G4;
#endif

#pragma unroll
    for (int q = 0; q < 2; ++q) {
      const int task = wave + q * 32;     // 0..63
      const int m = task >> 4;            // batch tile 0..3
      const int j = task & 15;            // hd tile 0..15
      const int col = j * 16 + lane15;    // hd channel

      v8f acc[4];
#pragma unroll
      for (int g = 0; g < 4; ++g) {
        const float bv = bias[g * HD + col];
#pragma unroll
        for (int r = 0; r < 8; ++r) {
          const int b = m * 16 + r + 8 * hf;
#if HAVE_TDM
          acc[g][r] = (float)gt[(size_t)b * G4 + g * HD + col] + bv;
#else
          acc[g][r] = (float)G[((size_t)b * T_SZ + t) * G4 + g * HD + col] + bv;
#endif
        }
      }
      for (int kk = 0; kk < HD; kk += 32) {
        v16bf a = make_frag(hbuf, HP, m * 16, kk);
#pragma unroll
        for (int g = 0; g < 4; ++g) {
          v16bf bf = make_frag(Whh, HD, g * HD + j * 16, kk);
          acc[g] = __builtin_amdgcn_wmma_f32_16x16x32_bf16(false, a, false, bf,
                                                           (short)0, acc[g], false, false);
        }
      }
      // pointwise gates: c = sig(f)*c + sig(i)*tanh(g); h = sig(o)*tanh(c)
#pragma unroll
      for (int r = 0; r < 8; ++r) {
        const int b = m * 16 + r + 8 * hf;
        const float iv = sigmoidf_(acc[0][r]);
        const float fv = sigmoidf_(acc[1][r]);
        const float gv = tanhf(acc[2][r]);
        const float ov = sigmoidf_(acc[3][r]);
        const float cn = fv * cbuf[b * HD + col] + iv * gv;
        cbuf[b * HD + col] = cn;
        const float hn = ov * tanhf(cn);
        hreg[q][r] = hn;
        h_out[((size_t)b * T_SZ + t) * H_SZ + dir * HD + col] = (bf16_t)hn;
      }
    }
    __syncthreads();   // all reads of old h done
#pragma unroll
    for (int q = 0; q < 2; ++q) {
      const int task = wave + q * 32;
      const int m = task >> 4, j = task & 15;
      const int col = j * 16 + lane15;
#pragma unroll
      for (int r = 0; r < 8; ++r)
        hbuf[(m * 16 + r + 8 * hf) * HP + col] = (bf16_t)hreg[q][r];
    }
    __syncthreads();   // new h visible to all waves
  }
}

// ---------------- tag projection: logits = (h_out @ W_tag^T + b) * mask ------
__global__ void k_tag_logits(const bf16_t* __restrict__ h_out,
                             const float* __restrict__ W_tag,
                             const float* __restrict__ b_tag,
                             const float* __restrict__ mask,
                             float* __restrict__ logits) {
  int bt = blockIdx.x * blockDim.x + threadIdx.x;
  if (bt >= BT) return;
  const bf16_t* hr = h_out + (size_t)bt * H_SZ;
  const float m = mask[bt];
  for (int k = 0; k < KT; ++k) {
    float a = b_tag[k];
    const float* w = W_tag + (size_t)k * H_SZ;
    for (int h = 0; h < H_SZ; ++h) a += (float)hr[h] * w[h];
    logits[(size_t)bt * KT + k] = a * m;
  }
}

// ---------------- CRF: forward partition Z + gold score per batch row --------
__global__ __launch_bounds__(32)
void k_crf(const float* __restrict__ logits, const float* __restrict__ mask,
           const int* __restrict__ label, const float* __restrict__ trans,
           float* __restrict__ loss_b) {
  const int b = blockIdx.x;
  const int k = threadIdx.x;
  __shared__ float trs[KT][KT];
  __shared__ float score[KT];
  __shared__ float tmp[KT];
  for (int i = k; i < KT * KT; i += 32) (&trs[0][0])[i] = trans[i];
  if (k < KT) score[k] = (k == START_TAG) ? 0.0f : NEGV;
  __syncthreads();

  for (int t = 0; t < T_SZ; ++t) {
    const float m = mask[b * T_SZ + t];
    float ns = 0.0f, old = 0.0f;
    if (k < KT) {
      old = score[k];
      const float hid = logits[((size_t)b * T_SZ + t) * KT + k];
      float mx = -3.4e38f;
      for (int j = 0; j < KT; ++j) mx = fmaxf(mx, score[j] + trs[k][j]);
      float sm = 0.0f;
      for (int j = 0; j < KT; ++j) sm += __expf(score[j] + trs[k][j] - mx);
      ns = mx + __logf(sm) + hid;
    }
    __syncthreads();
    if (k < KT) score[k] = ns * (1.0f - m) + old * m;
    __syncthreads();
  }
  if (k < KT) tmp[k] = score[k] + trans[END_TAG * KT + k];
  __syncthreads();
  if (k == 0) {
    float mx = -3.4e38f;
    for (int j = 0; j < KT; ++j) mx = fmaxf(mx, tmp[j]);
    float sm = 0.0f;
    for (int j = 0; j < KT; ++j) sm += __expf(tmp[j] - mx);
    const float Z = mx + __logf(sm);
    // gold path score
    float s = 0.0f, cnt = 0.0f;
    int prev = START_TAG;
    for (int t = 0; t < T_SZ; ++t) {
      const int   lbl = label[b * T_SZ + t];
      const float m   = mask[b * T_SZ + t];
      const float val = logits[((size_t)b * T_SZ + t) * KT + lbl] +
                        trans[lbl * KT + prev];
      s = val * (1.0f - m) + s * m;
      cnt += 1.0f - m;
      prev = lbl;
    }
    const int last = (int)(cnt - 1.0f);
    const int lastlbl = label[b * T_SZ + last];
    s += trans[END_TAG * KT + lastlbl];
    loss_b[b] = Z - s;
  }
}

__global__ void k_reduce_mean(const float* __restrict__ loss_b, float* out) {
  if (threadIdx.x == 0) {
    float s = 0.0f;
    for (int b = 0; b < B_SZ; ++b) s += loss_b[b];
    out[0] = s / (float)B_SZ;
  }
}

// ---------------- workspace layout ----------------
constexpr size_t alup(size_t x) { return (x + 255) & ~(size_t)255; }
constexpr size_t SZ_EMB  = (size_t)BT * E_SZ * 2;
constexpr size_t SZ_W    = (size_t)G4 * 256 * 2;     // Wih/Whh bf16
constexpr size_t SZ_BIAS = (size_t)G4 * 4;
constexpr size_t SZ_G    = (size_t)BT * G4 * 2;
constexpr size_t SZ_HOUT = (size_t)BT * H_SZ * 2;
constexpr size_t SZ_CWS  = (size_t)2 * B_SZ * HD * 4;
constexpr size_t SZ_LOG  = (size_t)BT * KT * 4;
constexpr size_t OFF_EMB   = 0;
constexpr size_t OFF_WIHF  = OFF_EMB  + alup(SZ_EMB);
constexpr size_t OFF_WHHF  = OFF_WIHF + alup(SZ_W);
constexpr size_t OFF_WIHB  = OFF_WHHF + alup(SZ_W);
constexpr size_t OFF_WHHB  = OFF_WIHB + alup(SZ_W);
constexpr size_t OFF_BIASF = OFF_WHHB + alup(SZ_W);
constexpr size_t OFF_BIASB = OFF_BIASF + alup(SZ_BIAS);
constexpr size_t OFF_GF    = OFF_BIASB + alup(SZ_BIAS);
constexpr size_t OFF_GB    = OFF_GF + alup(SZ_G);
constexpr size_t OFF_HOUT  = OFF_GB + alup(SZ_G);
constexpr size_t OFF_CWS   = OFF_HOUT + alup(SZ_HOUT);
constexpr size_t OFF_LOG   = OFF_CWS + alup(SZ_CWS);
constexpr size_t OFF_LOSSB = OFF_LOG + alup(SZ_LOG);

extern "C" void kernel_launch(void* const* d_in, const int* in_sizes, int n_in,
                              void* d_out, int out_size, void* d_ws, size_t ws_size,
                              hipStream_t stream) {
  (void)in_sizes; (void)n_in; (void)out_size; (void)ws_size;
  const int*   x      = (const int*)  d_in[0];
  const float* mask   = (const float*)d_in[1];
  const int*   label  = (const int*)  d_in[2];
  const float* embed  = (const float*)d_in[3];
  const float* Wih_f  = (const float*)d_in[4];
  const float* Whh_f  = (const float*)d_in[5];
  const float* bih_f  = (const float*)d_in[6];
  const float* bhh_f  = (const float*)d_in[7];
  const float* Wih_b  = (const float*)d_in[8];
  const float* Whh_b  = (const float*)d_in[9];
  const float* bih_b  = (const float*)d_in[10];
  const float* bhh_b  = (const float*)d_in[11];
  const float* W_tag  = (const float*)d_in[12];
  const float* b_tag  = (const float*)d_in[13];
  const float* trans  = (const float*)d_in[14];
  float* out = (float*)d_out;

  char* ws = (char*)d_ws;
  bf16_t* emb_bf   = (bf16_t*)(ws + OFF_EMB);
  bf16_t* wih_f_bf = (bf16_t*)(ws + OFF_WIHF);
  bf16_t* whh_f_bf = (bf16_t*)(ws + OFF_WHHF);
  bf16_t* wih_b_bf = (bf16_t*)(ws + OFF_WIHB);
  bf16_t* whh_b_bf = (bf16_t*)(ws + OFF_WHHB);
  float*  bias_f   = (float*) (ws + OFF_BIASF);
  float*  bias_b   = (float*) (ws + OFF_BIASB);
  bf16_t* G_f      = (bf16_t*)(ws + OFF_GF);
  bf16_t* G_b      = (bf16_t*)(ws + OFF_GB);
  bf16_t* h_out    = (bf16_t*)(ws + OFF_HOUT);
  float*  c_ws     = (float*) (ws + OFF_CWS);
  float*  logits   = (float*) (ws + OFF_LOG);
  float*  loss_b   = (float*) (ws + OFF_LOSSB);

  // 1) prep: embed gather (fp32 -> bf16), weight converts, fused biases
  {
    size_t n = (size_t)BT * E_SZ;
    k_embed_gather<<<(unsigned)((n + 255) / 256), 256, 0, stream>>>(x, embed, emb_bf);
    const int nw = G4 * 256;
    k_f32_to_bf16<<<(nw + 255) / 256, 256, 0, stream>>>(Wih_f, wih_f_bf, nw);
    k_f32_to_bf16<<<(nw + 255) / 256, 256, 0, stream>>>(Whh_f, whh_f_bf, nw);
    k_f32_to_bf16<<<(nw + 255) / 256, 256, 0, stream>>>(Wih_b, wih_b_bf, nw);
    k_f32_to_bf16<<<(nw + 255) / 256, 256, 0, stream>>>(Whh_b, whh_b_bf, nw);
    k_bias_fuse<<<(G4 + 255) / 256, 256, 0, stream>>>(bih_f, bhh_f, bias_f, G4);
    k_bias_fuse<<<(G4 + 255) / 256, 256, 0, stream>>>(bih_b, bhh_b, bias_b, G4);
  }
  // 2) device-wide WMMA GEMMs: G = emb @ Wih^T  (both directions)
  {
    dim3 grid(BT / 128, G4 / 64);
    k_gemm_xproj<<<grid, 256, 0, stream>>>(emb_bf, wih_f_bf, G_f);
    k_gemm_xproj<<<grid, 256, 0, stream>>>(emb_bf, wih_b_bf, G_b);
  }
  // 3) persistent recurrent LSTM, one workgroup per direction,
  //    TDM double-buffered G tiles in dynamic LDS (~289 KB of the WGP's 320 KB)
  k_lstm_recurrent<<<2, 1024, (unsigned)SM_RECUR, stream>>>(
      G_f, G_b, whh_f_bf, whh_b_bf, bias_f, bias_b, c_ws, h_out);
  // 4) tag projection + mask
  k_tag_logits<<<BT / 256, 256, 0, stream>>>(h_out, W_tag, b_tag, mask, logits);
  // 5) CRF forward + gold score, per batch row
  k_crf<<<B_SZ, 32, 0, stream>>>(logits, mask, label, trans, loss_b);
  // 6) mean over batch -> scalar loss
  k_reduce_mean<<<1, 32, 0, stream>>>(loss_b, out);
}
